// GraphPredGCN_50586124812352
// MI455X (gfx1250) — matrix operand order repaired and checked by reading, not compiled, not run on hardware
//
#include <hip/hip_runtime.h>
#include <hip/hip_bf16.h>

typedef __bf16 bf16;
typedef __attribute__((ext_vector_type(16))) __bf16 v16bf;
typedef __attribute__((ext_vector_type(8)))  __bf16 v8bf;
typedef __attribute__((ext_vector_type(4)))  __bf16 v4bf;
typedef __attribute__((ext_vector_type(8)))  float  v8f;

#define N_NODES  100000
#define N_EDGES  1600000
#define N_GRAPHS 512
#define HID      128
#define N_CLS    10

// ---------------- precision conversion (float4 -> 4x bf16, b64 store) ----------------
__global__ void k_cvt_x(const float* __restrict__ x, bf16* __restrict__ hbf, int n4) {
  int i = blockIdx.x * blockDim.x + threadIdx.x;
  if (i >= n4) return;
  float4 v = ((const float4*)x)[i];
  v4bf o; o[0] = (bf16)v.x; o[1] = (bf16)v.y; o[2] = (bf16)v.z; o[3] = (bf16)v.w;
  ((v4bf*)hbf)[i] = o;
}

// Wt[n][k] = W[k][n]  (transposed bf16 so B fragments are contiguous in K)
__global__ void k_cvt_w(const float* __restrict__ W, bf16* __restrict__ Wt) {
  int i = blockIdx.x * blockDim.x + threadIdx.x;
  if (i >= HID * HID) return;
  int n = i >> 7, k = i & 127;
  Wt[n * HID + k] = (bf16)W[k * HID + n];
}

// ---------------- degree / normalization ----------------
__global__ void k_deg_init(float* __restrict__ deg) {
  int i = blockIdx.x * blockDim.x + threadIdx.x;
  if (i < N_NODES) deg[i] = 1.0f;            // self-loop
}

__global__ void k_deg_edges(const int* __restrict__ ei, float* __restrict__ deg) {
  int e = blockIdx.x * blockDim.x + threadIdx.x;
  if (e < N_EDGES) unsafeAtomicAdd(&deg[ei[N_EDGES + e]], 1.0f);  // col = target
}

__global__ void k_dinv(float* __restrict__ deg) {
  int i = blockIdx.x * blockDim.x + threadIdx.x;
  if (i < N_NODES) deg[i] = rsqrtf(fmaxf(deg[i], 1.0f));
}

// ---------------- WMMA GEMM: C[M,128] = A[M,128](bf16) @ Wt^T (bf16), f32 accum ----------------
__global__ __launch_bounds__(256) void k_gemm(const bf16* __restrict__ A,
                                              const bf16* __restrict__ Bt,  // [n][k] bf16
                                              float* __restrict__ C, int M) {
  __shared__ bf16 lds_w[HID * HID];          // 32 KB, whole weight matrix
  const uint4* src = (const uint4*)Bt;
  uint4* dst = (uint4*)lds_w;
  int tid = threadIdx.x;
#pragma unroll
  for (int i = 0; i < 8; ++i) dst[tid + 256 * i] = src[tid + 256 * i];   // 2048 x 16B
  __syncthreads();

  int wave = tid >> 5;
  int lane = tid & 31;
  int m0 = (blockIdx.x * 8 + wave) * 16;
  if (m0 >= M) return;

  int mrow = m0 + (lane & 15);
  int hi   = lane >> 4;                      // 0: lanes 0-15, 1: lanes 16-31

  // A fragments: 4 k-steps of 32. Per ISA 16-bit A layout:
  // lanes 0-15 hold K = {k0..k0+7, k0+16..k0+23}, lanes 16-31 hold +8 of each.
  v16bf a[4];
#pragma unroll
  for (int kk = 0; kk < 4; ++kk) {
    int kb = kk * 32 + hi * 8;
    v8bf lo = *(const v8bf*)(A + mrow * HID + kb);
    v8bf hv = *(const v8bf*)(A + mrow * HID + kb + 16);
#pragma unroll
    for (int j = 0; j < 8; ++j) { a[kk][j] = lo[j]; a[kk][j + 8] = hv[j]; }
  }

#pragma unroll
  for (int nt = 0; nt < 8; ++nt) {
    // B fragment: lane = column n, same K split as A (lanes 16-31 take +8 of each 8-run)
    int n = nt * 16 + (lane & 15);
    v8f acc = {};
#pragma unroll
    for (int kk = 0; kk < 4; ++kk) {
      v8bf blo = *(const v8bf*)(lds_w + n * HID + kk * 32 + hi * 8);
      v8bf bhi = *(const v8bf*)(lds_w + n * HID + kk * 32 + hi * 8 + 16);
      v16bf b;
#pragma unroll
      for (int j = 0; j < 8; ++j) { b[j] = blo[j]; b[j + 8] = bhi[j]; }
      acc = __builtin_amdgcn_wmma_f32_16x16x32_bf16(
          false, a[kk], false, b, (short)0, acc, false, false);
    }
    // D layout: VGPR r -> row m0 + r + 8*hi, col = n0 + (lane&15)
    int col = nt * 16 + (lane & 15);
    int rb  = m0 + hi * 8;
#pragma unroll
    for (int r = 0; r < 8; ++r) C[(rb + r) * HID + col] = acc[r];
  }
}

// ---------------- aggregation ----------------
// agg = self-loop term + bias (no atomics needed for the diagonal); float4 wide
__global__ void k_init_agg(const float* __restrict__ t, const float* __restrict__ dinv,
                           const float* __restrict__ b, float* __restrict__ agg, int n4) {
  int i = blockIdx.x * blockDim.x + threadIdx.x;
  if (i >= n4) return;
  float dv = dinv[i >> 5];
  float s = dv * dv;
  float4 tv = ((const float4*)t)[i];
  float4 bv = ((const float4*)b)[i & 31];
  float4 o;
  o.x = tv.x * s + bv.x; o.y = tv.y * s + bv.y;
  o.z = tv.z * s + bv.z; o.w = tv.w * s + bv.w;
  ((float4*)agg)[i] = o;
}

// edge scatter: one wave per edge, float4 per lane.
// agg[col] += t[row] * dinv[row]*dinv[col]
__global__ void k_edge_agg(const int* __restrict__ ei, const float* __restrict__ dinv,
                           const float* __restrict__ t, float* __restrict__ agg) {
  long long id = (long long)blockIdx.x * blockDim.x + threadIdx.x;
  if (id >= (long long)N_EDGES * 32) return;
  int e  = (int)(id >> 5);
  int f4 = (int)(id & 31) << 2;
  int r = ei[e];
  int c = ei[N_EDGES + e];
  float nv = dinv[r] * dinv[c];
  float4 v = *(const float4*)(t + r * HID + f4);           // global_load_b128 gather
  float* dp = agg + c * HID + f4;
  unsafeAtomicAdd(dp + 0, v.x * nv);
  unsafeAtomicAdd(dp + 1, v.y * nv);
  unsafeAtomicAdd(dp + 2, v.z * nv);
  unsafeAtomicAdd(dp + 3, v.w * nv);
}

// ReLU + requantize for next layer; keep f32 copy in hout (reuses t buffer); float4 wide
__global__ void k_finalize(const float* __restrict__ agg, bf16* __restrict__ hbf,
                           float* __restrict__ hout, int n4) {
  int i = blockIdx.x * blockDim.x + threadIdx.x;
  if (i >= n4) return;
  float4 v = ((const float4*)agg)[i];
  v.x = fmaxf(v.x, 0.0f); v.y = fmaxf(v.y, 0.0f);
  v.z = fmaxf(v.z, 0.0f); v.w = fmaxf(v.w, 0.0f);
  v4bf o; o[0] = (bf16)v.x; o[1] = (bf16)v.y; o[2] = (bf16)v.z; o[3] = (bf16)v.w;
  ((v4bf*)hbf)[i] = o;                                      // b64 store
  ((float4*)hout)[i] = v;                                   // b128 store
}

// ---------------- pooling + head ----------------
__global__ void k_pool_zero(float* __restrict__ pooled, float* __restrict__ cnt) {
  int i = blockIdx.x * blockDim.x + threadIdx.x;
  if (i < N_GRAPHS * HID) pooled[i] = 0.0f;
  else if (i < N_GRAPHS * HID + N_GRAPHS) cnt[i - N_GRAPHS * HID] = 0.0f;
}

__global__ void k_pool(const int* __restrict__ batch, const float* __restrict__ h,
                       float* __restrict__ pooled, float* __restrict__ cnt) {
  int i = blockIdx.x * blockDim.x + threadIdx.x;
  if (i >= N_NODES * 32) return;
  int node = i >> 5, f4 = (i & 31) << 2;
  int g = batch[node];
  float4 v = *(const float4*)(h + node * HID + f4);
  float* dp = pooled + g * HID + f4;
  unsafeAtomicAdd(dp + 0, v.x);
  unsafeAtomicAdd(dp + 1, v.y);
  unsafeAtomicAdd(dp + 2, v.z);
  unsafeAtomicAdd(dp + 3, v.w);
  if (f4 == 0) unsafeAtomicAdd(&cnt[g], 1.0f);
}

__global__ __launch_bounds__(128) void k_head(const float* __restrict__ pooled,
                                              const float* __restrict__ cnt,
                                              const float* __restrict__ Wl,
                                              const float* __restrict__ bl,
                                              float* __restrict__ out) {
  int g = blockIdx.x, t = threadIdx.x;
  __shared__ float acc[N_CLS];
  if (t < N_CLS) acc[t] = 0.0f;
  __syncthreads();
  float m = pooled[g * HID + t] / fmaxf(cnt[g], 1.0f);
#pragma unroll
  for (int c = 0; c < N_CLS; ++c) atomicAdd(&acc[c], m * Wl[t * N_CLS + c]);
  __syncthreads();
  if (t < N_CLS) out[g * N_CLS + t] = acc[t] + bl[t];
}

// ---------------- launcher ----------------
static inline size_t align_up(size_t x, size_t a) { return (x + a - 1) & ~(a - 1); }

extern "C" void kernel_launch(void* const* d_in, const int* in_sizes, int n_in,
                              void* d_out, int out_size, void* d_ws, size_t ws_size,
                              hipStream_t stream) {
  const float* x     = (const float*)d_in[0];
  const int*   ei    = (const int*)d_in[1];
  const int*   batch = (const int*)d_in[2];
  const float* W0    = (const float*)d_in[3];
  const float* b0    = (const float*)d_in[4];
  const float* W1    = (const float*)d_in[5];
  const float* b1    = (const float*)d_in[6];
  const float* W2    = (const float*)d_in[7];
  const float* b2    = (const float*)d_in[8];
  const float* Wl    = (const float*)d_in[9];
  const float* bl    = (const float*)d_in[10];
  float* out = (float*)d_out;

  // workspace carve-out
  char* ws = (char*)d_ws;
  size_t o = 0;
  bf16*  hbf    = (bf16*)(ws + o);  o = align_up(o + (size_t)N_NODES * HID * sizeof(bf16), 256);
  bf16*  wbf    = (bf16*)(ws + o);  o = align_up(o + (size_t)3 * HID * HID * sizeof(bf16), 256);
  float* t      = (float*)(ws + o); o = align_up(o + (size_t)N_NODES * HID * sizeof(float), 256);
  float* agg    = (float*)(ws + o); o = align_up(o + (size_t)N_NODES * HID * sizeof(float), 256);
  float* dinv   = (float*)(ws + o); o = align_up(o + (size_t)N_NODES * sizeof(float), 256);
  float* pooled = (float*)(ws + o); o = align_up(o + (size_t)N_GRAPHS * HID * sizeof(float), 256);
  float* cnt    = (float*)(ws + o); o = align_up(o + (size_t)N_GRAPHS * sizeof(float), 256);
  (void)ws_size; (void)n_in; (void)in_sizes; (void)out_size;

  const int NB = 256;
  const int nh4 = N_NODES * HID / 4;   // float4 element count

  // precision prep
  k_cvt_x<<<(nh4 + NB - 1) / NB, NB, 0, stream>>>(x, hbf, nh4);
  k_cvt_w<<<(HID * HID + NB - 1) / NB, NB, 0, stream>>>(W0, wbf + 0 * HID * HID);
  k_cvt_w<<<(HID * HID + NB - 1) / NB, NB, 0, stream>>>(W1, wbf + 1 * HID * HID);
  k_cvt_w<<<(HID * HID + NB - 1) / NB, NB, 0, stream>>>(W2, wbf + 2 * HID * HID);

  // degrees -> dinv
  k_deg_init<<<(N_NODES + NB - 1) / NB, NB, 0, stream>>>(dinv);
  k_deg_edges<<<(N_EDGES + NB - 1) / NB, NB, 0, stream>>>(ei, dinv);
  k_dinv<<<(N_NODES + NB - 1) / NB, NB, 0, stream>>>(dinv);

  const int gemm_blocks = (N_NODES / 16 + 7) / 8;                 // 8 waves * 16 rows / block
  const long long et = (long long)N_EDGES * 32;                   // wave per edge, float4/lane
  const int edge_blocks = (int)((et + NB - 1) / NB);
  const float* biases[3] = {b0, b1, b2};

  for (int l = 0; l < 3; ++l) {
    k_gemm<<<gemm_blocks, 256, 0, stream>>>(hbf, wbf + (size_t)l * HID * HID, t, N_NODES);
    k_init_agg<<<(nh4 + NB - 1) / NB, NB, 0, stream>>>(t, dinv, biases[l], agg, nh4);
    k_edge_agg<<<edge_blocks, NB, 0, stream>>>(ei, dinv, t, agg);
    k_finalize<<<(nh4 + NB - 1) / NB, NB, 0, stream>>>(agg, hbf, t, nh4);
  }

  // pooling + classifier head
  k_pool_zero<<<(N_GRAPHS * (HID + 1) + NB - 1) / NB, NB, 0, stream>>>(pooled, cnt);
  k_pool<<<(N_NODES * 32 + NB - 1) / NB, NB, 0, stream>>>(batch, t, pooled, cnt);
  k_head<<<N_GRAPHS, 128, 0, stream>>>(pooled, cnt, Wl, bl, out);
}
